// DLI_loss_full_6614249636379
// MI455X (gfx1250) — compile-verified, matrix operand order checked by reading, not run on hardware
//
#include <hip/hip_runtime.h>
#include <hip/hip_bf16.h>
#include <math.h>

// ---- problem constants -----------------------------------------------------
#define BB   32
#define SS   1024
#define TT   64
#define ENC  768
#define HID  256
#define G4   1024          // 4*HID
#define BT   2048          // BB*TT

typedef __attribute__((ext_vector_type(16))) __bf16 v16bf;
typedef __attribute__((ext_vector_type(8)))  float  v8f;
typedef __attribute__((ext_vector_type(4)))  unsigned v4u;
typedef __attribute__((ext_vector_type(8)))  int      v8i;
typedef __attribute__((ext_vector_type(4)))  int      v4i;

// K index inside a 16-element (8 VGPR) bf16 A/B fragment, per CDNA5 ISA:
// VGPR v holds K = (v/4)*16 + half*8 + (v%4)*2 + p  (contiguous 16B chunks,
// which the compiler merges into b128 loads)
__device__ __forceinline__ int kidx(int i, int half) {
    int v = i >> 1, p = i & 1;
    return ((v >> 2) << 4) + (half << 3) + ((v & 3) << 1) + p;
}

__device__ __forceinline__ float sigf(float x) { return 1.0f / (1.0f + expf(-x)); }

// ---- K1: weight conversion + bias fold + barrier reset ---------------------
__global__ void k_convert(const float* __restrict__ Wih, const float* __restrict__ Whh,
                          const float* __restrict__ bih, const float* __restrict__ bhh,
                          __bf16* __restrict__ Wihb, __bf16* __restrict__ Whhb,
                          float* __restrict__ bias, unsigned* __restrict__ bar) {
    int i = blockIdx.x * blockDim.x + threadIdx.x;
    if (i < G4 * ENC) Wihb[i] = (__bf16)Wih[i];
    if (i < G4 * HID) Whhb[i] = (__bf16)Whh[i];
    if (i < G4)       bias[i] = bih[i] + bhh[i];
    if (i == 0)       bar[0] = 0u;
}

// ---- K2: gather turn states (masked) -> bf16, and e = TS . w_e ------------
__global__ void k_gather(const float* __restrict__ enc, const int* __restrict__ ids,
                         const int* __restrict__ lens, const float* __restrict__ fcw,
                         __bf16* __restrict__ TS, float* __restrict__ e_arr) {
    int m = blockIdx.x;            // 0..BT-1
    int b = m >> 6, t = m & 63;
    int tid = threadIdx.x;         // 256 threads
    int src = ids[b * TT + t];
    bool valid = t < lens[b];
    const float* row = enc + ((size_t)b * SS + src) * ENC;
    float partial = 0.0f;
    for (int i = tid; i < ENC; i += 256) {
        float v = valid ? row[i] : 0.0f;
        TS[(size_t)m * ENC + i] = (__bf16)v;
        partial += v * fcw[HID + i];          // w_e = fc_w[0, HID:]
    }
    __shared__ float red[256];
    red[tid] = partial; __syncthreads();
    for (int s = 128; s > 0; s >>= 1) { if (tid < s) red[tid] += red[tid + s]; __syncthreads(); }
    if (tid == 0) e_arr[m] = red[0];
}

// ---- K3: Xg[2048,1024] = TS @ W_ih^T + bias  (bf16 WMMA, f32 accum) -------
// per-wave tile: 32 (M) x 64 (N): every B fragment feeds 2 WMMAs, halving
// W_ih L2 traffic vs a 16-row tile. 1024 waves total.
__global__ void k_gemm_xg(const __bf16* __restrict__ TS, const __bf16* __restrict__ Wihb,
                          const float* __restrict__ bias, float* __restrict__ Xg) {
    int gw   = blockIdx.x * 8 + (threadIdx.x >> 5);   // global wave id, 0..1023
    int lane = threadIdx.x & 31;
    int lr = lane & 15, half = lane >> 4;
    int mp = gw >> 4;                                  // 0..63 (rows mp*32..+31)
    int nq = gw & 15;                                  // 0..15 (4 n-tiles each)
    v8f acc[2][4] = {};
    for (int kk = 0; kk < 24; ++kk) {
        int k0 = kk * 32;
        const __bf16* ap0 = TS + ((size_t)(mp * 32)      + lr) * ENC + k0;
        const __bf16* ap1 = TS + ((size_t)(mp * 32 + 16) + lr) * ENC + k0;
        v16bf A0, A1;
        #pragma unroll
        for (int i = 0; i < 16; ++i) {
            int o = kidx(i, half);
            A0[i] = ap0[o];
            A1[i] = ap1[o];
        }
        #pragma unroll
        for (int j = 0; j < 4; ++j) {
            int ncol = (nq * 4 + j) * 16 + lr;
            const __bf16* bp = Wihb + (size_t)ncol * ENC + k0;   // B[k,n] = W_ih[n,k]
            v16bf Bv;
            #pragma unroll
            for (int i = 0; i < 16; ++i) Bv[i] = bp[kidx(i, half)];
            acc[0][j] = __builtin_amdgcn_wmma_f32_16x16x32_bf16(
                false, A0, false, Bv, (short)0, acc[0][j], false, false);
            acc[1][j] = __builtin_amdgcn_wmma_f32_16x16x32_bf16(
                false, A1, false, Bv, (short)0, acc[1][j], false, false);
        }
    }
    #pragma unroll
    for (int mi = 0; mi < 2; ++mi) {
        #pragma unroll
        for (int j = 0; j < 4; ++j) {
            int n = (nq * 4 + j) * 16 + lr;
            float bn = bias[n];
            #pragma unroll
            for (int r = 0; r < 8; ++r) {
                int m = mp * 32 + mi * 16 + r + half * 8;
                Xg[(size_t)m * G4 + n] = acc[mi][j][r] + bn;
            }
        }
    }
}

// ---- K4: 4-WGP recurrent LSTM ---------------------------------------------
// Block q (of 4) owns gate q = output columns [256q, 256q+256); its 128KB
// W_hh quarter is DMA'd into LDS once by the Tensor Data Mover and stays
// resident for all 64 steps. Per step: LDS-fed WMMA GEMM -> gate quarter to
// double-buffered global Gbuf -> device barrier -> redundant elementwise
// cell on every block (h, c stay local; h -> LDS bf16 next-step A operand).
__global__ void __launch_bounds__(1024) k_lstm4(const __bf16* __restrict__ Whhb,
                                                const float* __restrict__ Xg,
                                                const float* __restrict__ fcw,
                                                float* __restrict__ a_arr,
                                                float* __restrict__ Gbuf,   // [2][4][32][256]
                                                unsigned* __restrict__ bar) {
    extern __shared__ char smem[];
    __bf16* Wlds = (__bf16*)smem;                       // 256x256 bf16 = 128 KB
    __bf16* hbf  = (__bf16*)(smem + HID * HID * 2);     // 32x256  bf16 = 16 KB

    int q   = blockIdx.x;          // gate / N-quarter
    int tid = threadIdx.x;
    int w = tid >> 5, lane = tid & 31;
    int lr = lane & 15, half = lane >> 4;

    // --- TDM: DMA the W_hh quarter (128 KB contiguous) into LDS, once -----
    if (tid < 32) {                // wave 0 issues; TENSORcnt is per-wave
        unsigned ldsoff = (unsigned)__builtin_amdgcn_groupstaticsize(); // dyn LDS base
        unsigned long long ga =
            (unsigned long long)(const void*)(Whhb + (size_t)q * HID * HID);
        // D# group0: count=1 | lds_addr | global_addr[56:0] | type=2
        v4u g0 = { 1u,
                   ldsoff,
                   (unsigned)(ga & 0xFFFFFFFFu),
                   (unsigned)(((ga >> 32) & 0x1FFFFFFu) | (2u << 30)) };
        // D# group1: data_size=3 (8B), tensor_dim0=tile_dim0=16384 elems,
        // tensor_dim1=1, dim0_stride=16384 (1-D tile = whole 128 KB block)
        v8i g1 = { (int)(3u << 16),
                   (int)(0x4000u << 16),      // tensor_dim0[15:0] << 16
                   (int)(1u << 16),           // tensor_dim0[31:16]=0 | tensor_dim1[15:0]=1
                   (int)(0x4000u << 16),      // tensor_dim1[31:16]=0 | tile_dim0=16384
                   0,                         // tile_dim1=0 (unused), tile_dim2=0
                   16384,                     // tensor_dim0_stride[31:0]
                   0, 0 };
        v4i g2 = { 1, 1, 0, 0 };              // tensor_dim2=1, tensor_dim3=1
        v4i g3 = { 0, 0, 0, 0 };
        v8i g4 = { 0, 0, 0, 0, 0, 0, 0, 0 };  // unused trailing group (6-arg form)
        __builtin_amdgcn_tensor_load_to_lds(g0, g1, g2, g3, g4, 0);
        __builtin_amdgcn_s_wait_tensorcnt(0);
    }
    for (int i = tid; i < BB * HID; i += 1024) hbf[i] = (__bf16)0.0f;
    float cst[8];
    #pragma unroll
    for (int u = 0; u < 8; ++u) cst[u] = 0.0f;
    __syncthreads();

    int mt = w & 1;                // M-tile (batches 0-15 / 16-31)
    int nt = w >> 1;               // N-tile 0..15 within the quarter

    for (int t = 0; t < TT; ++t) {
        int slot = t & 1;
        float* Gs = Gbuf + (size_t)slot * 4 * BB * HID;

        // --- recurrent GEMM: one 16x16 tile per wave, K = 256 --------------
        v8f acc = {};
        for (int kk = 0; kk < 8; ++kk) {
            int k0 = kk * 32;
            v16bf A, Bv;
            #pragma unroll
            for (int i = 0; i < 16; ++i) {
                int k = k0 + kidx(i, half);
                A[i]  = hbf[(mt * 16 + lr) * HID + k];
                Bv[i] = Wlds[(nt * 16 + lr) * HID + k];   // B[k,n] = W_hh[n,k]
            }
            acc = __builtin_amdgcn_wmma_f32_16x16x32_bf16(
                false, A, false, Bv, (short)0, acc, false, false);
        }
        int nl = nt * 16 + lr;     // local column within quarter
        #pragma unroll
        for (int r = 0; r < 8; ++r) {
            int b = mt * 16 + r + half * 8;
            Gs[(q * BB + b) * HID + nl] =
                acc[r] + Xg[((size_t)b * TT + t) * G4 + q * HID + nl];
        }

        // --- device barrier: all four gate quarters visible ----------------
        __threadfence();
        __syncthreads();
        if (tid == 0) {
            __hip_atomic_fetch_add(bar, 1u, __ATOMIC_ACQ_REL, __HIP_MEMORY_SCOPE_AGENT);
            unsigned target = 4u * (unsigned)(t + 1);
            while (__hip_atomic_load(bar, __ATOMIC_ACQUIRE, __HIP_MEMORY_SCOPE_AGENT) < target)
                __builtin_amdgcn_s_sleep(1);
        }
        __syncthreads();

        // --- LSTM cell (redundant on all 4 blocks): wave w == batch b ------
        float partial = 0.0f;
        #pragma unroll
        for (int u = 0; u < 8; ++u) {
            int j  = lane * 8 + u;
            float ig = Gs[(0 * BB + w) * HID + j];
            float fg = Gs[(1 * BB + w) * HID + j];
            float gg = Gs[(2 * BB + w) * HID + j];
            float og = Gs[(3 * BB + w) * HID + j];
            float cv = sigf(fg) * cst[u] + sigf(ig) * tanhf(gg);
            cst[u] = cv;
            float hv = sigf(og) * tanhf(cv);
            hbf[w * HID + j] = (__bf16)hv;
            partial += hv * fcw[j];            // w_h = fc_w[0, :HID]
        }
        #pragma unroll
        for (int off = 16; off > 0; off >>= 1)
            partial += __shfl_xor(partial, off, 32);
        if (q == 0 && lane == 0) a_arr[w * TT + t] = partial;
        __syncthreads();                        // hbf ready before next GEMM
    }
}

// ---- K5: per-batch loss (masked suffix logsumexp minus next-score) --------
__global__ void k_loss(const float* __restrict__ a_arr, const float* __restrict__ e_arr,
                       const float* __restrict__ fcb, const int* __restrict__ lens,
                       float* __restrict__ num, float* __restrict__ cnt) {
    int b = blockIdx.x;
    int j = threadIdx.x;  // 0..63
    __shared__ float e_s[TT];
    __shared__ float red[TT];
    e_s[j] = e_arr[b * TT + j];
    __syncthreads();
    int L = lens[b];
    float fb = fcb[0];
    float lj = 0.0f;
    if (j < L - 1 && j < TT - 1) {
        float m = -1e30f;
        for (int k = j + 1; k < L; ++k) m = fmaxf(m, e_s[k]);
        float s = 0.0f;
        for (int k = j + 1; k < L; ++k) s += expf(e_s[k] - m);
        float lse = a_arr[b * TT + j] + fb + m + logf(s);
        float sn  = a_arr[b * TT + j] + e_s[j + 1] + fb;
        lj = lse - sn;
    }
    red[j] = lj; __syncthreads();
    for (int s = 32; s > 0; s >>= 1) { if (j < s) red[j] += red[j + s]; __syncthreads(); }
    if (j == 0) { num[b] = red[0]; cnt[b] = (float)(L - 1); }
}

// ---- K6: final scalar ------------------------------------------------------
__global__ void k_final(const float* __restrict__ num, const float* __restrict__ cnt,
                        float* __restrict__ out) {
    if (threadIdx.x == 0 && blockIdx.x == 0) {
        float n = 0.0f, d = 0.0f;
        for (int i = 0; i < BB; ++i) { n += num[i]; d += cnt[i]; }
        out[0] = n / d;
    }
}

// ---- launcher --------------------------------------------------------------
extern "C" void kernel_launch(void* const* d_in, const int* in_sizes, int n_in,
                              void* d_out, int out_size, void* d_ws, size_t ws_size,
                              hipStream_t stream) {
    const float* enc  = (const float*)d_in[0];
    const float* Wih  = (const float*)d_in[1];
    const float* Whh  = (const float*)d_in[2];
    const float* bih  = (const float*)d_in[3];
    const float* bhh  = (const float*)d_in[4];
    const float* fcw  = (const float*)d_in[5];
    const float* fcb  = (const float*)d_in[6];
    const int*   ids  = (const int*)d_in[7];
    const int*   lens = (const int*)d_in[8];

    char* ws = (char*)d_ws;
    __bf16* TSb   = (__bf16*)ws;   ws += (size_t)BT * ENC * 2;         // 3.0 MB
    __bf16* Wihb  = (__bf16*)ws;   ws += (size_t)G4 * ENC * 2;         // 1.5 MB
    __bf16* Whhb  = (__bf16*)ws;   ws += (size_t)G4 * HID * 2;         // 0.5 MB
    float*  bias  = (float*)ws;    ws += (size_t)G4 * 4;
    float*  Xg    = (float*)ws;    ws += (size_t)BT * G4 * 4;          // 8.0 MB
    float*  Gbuf  = (float*)ws;    ws += (size_t)2 * 4 * BB * HID * 4; // 256 KB
    float*  e_arr = (float*)ws;    ws += (size_t)BT * 4;
    float*  a_arr = (float*)ws;    ws += (size_t)BT * 4;
    float*  num   = (float*)ws;    ws += (size_t)BB * 4;
    float*  cnt   = (float*)ws;    ws += (size_t)BB * 4;
    unsigned* bar = (unsigned*)ws; ws += 256;

    const int lstm_lds = HID * HID * 2 + BB * HID * 2;  // 128KB + 16KB (<320KB WGP LDS)
    (void)hipFuncSetAttribute((const void*)k_lstm4,
                              hipFuncAttributeMaxDynamicSharedMemorySize, lstm_lds);

    k_convert<<<(G4 * ENC + 255) / 256, 256, 0, stream>>>(Wih, Whh, bih, bhh,
                                                          Wihb, Whhb, bias, bar);
    k_gather<<<BT, 256, 0, stream>>>(enc, ids, lens, fcw, TSb, e_arr);
    k_gemm_xg<<<128, 256, 0, stream>>>(TSb, Wihb, bias, Xg);          // 1024 waves
    k_lstm4<<<4, 1024, lstm_lds, stream>>>(Whhb, Xg, fcw, a_arr, Gbuf, bar);
    k_loss<<<BB, TT, 0, stream>>>(a_arr, e_arr, fcb, lens, num, cnt);
    k_final<<<1, 1, 0, stream>>>(num, cnt, (float*)d_out);
}
